// EdgeNodePosConv_83288005804813
// MI455X (gfx1250) — matrix operand order repaired and checked by reading, not compiled, not run on hardware
//
#include <hip/hip_runtime.h>
#include <hip/hip_bf16.h>
#include <math.h>

// ---------------------------------------------------------------------------
// MI455X / gfx1250 fused GNN block.
// All matmuls through v_wmma_f32_16x16x32_f16 (wave32, f32 accumulate).
// Weights prepacked once to f16 B-fragment layout in d_ws (L2-resident).
// h and edge_type are cached as f16 in d_ws so the hot edge kernel stages
// tiles with b128 load/store pairs only (no per-edge f32->f16 conversion).
// ---------------------------------------------------------------------------

typedef __attribute__((ext_vector_type(16))) _Float16 v16h;
typedef __attribute__((ext_vector_type(8)))  float    v8f;
typedef _Float16 half_t;

#define NN 20000
#define EE 320000

// ---- fast math helpers (hardware v_rcp_f32 / v_rsq_f32, no IEEE div) -------

__device__ __forceinline__ float fast_silu(float v) {
  return v * __builtin_amdgcn_rcpf(1.f + __expf(-v));
}
__device__ __forceinline__ float fast_rsqrt(float v) {
  return __builtin_amdgcn_rsqf(v);
}

// ---- fragment helpers ------------------------------------------------------

__device__ __forceinline__ v16h ld_bfrag(const half_t* p) {
  return *(const v16h*)p;            // 32B contiguous, 32B aligned
}

// pack two f32 -> half2 and store as one b32
__device__ __forceinline__ void st_half2(half_t* p, float a, float b) {
  union { _Float16 h[2]; unsigned u; } t;
  t.h[0] = (_Float16)a; t.h[1] = (_Float16)b;
  *(unsigned*)p = t.u;
}
__device__ __forceinline__ unsigned pack_half2(float a, float b) {
  union { _Float16 h[2]; unsigned u; } t;
  t.h[0] = (_Float16)a; t.h[1] = (_Float16)b;
  return t.u;
}

// One fused MLP "layer tile": D(16xNT*16) = silu?(A(16xKT*32) x W + b)
// src: LDS activation tile (f16 row-major, strideH halves per row)
// dst: LDS destination (f16), or nullptr -> leave fp32 accumulators in accOut
// fragW: global prepacked B fragments, indexed [kt*NT + nt], 512 halves each.
// Even NT is processed two N-tiles at a time -> two independent WMMA chains.
template<int KT, int NT>
__device__ __forceinline__ void mlp_layer(
    const half_t* __restrict__ src, int strideH,
    half_t* __restrict__ dst,
    const half_t* __restrict__ fragW,
    const float* __restrict__ bias, int nBound,
    int lane, bool doSilu, v8f* accOut)
{
  const int nl = lane & 15, hf = lane >> 4;
  // Load all A fragments first (row M = lane&15; contiguous k = 16*hf + s, as in prepack).
  v16h a[KT];
#pragma unroll
  for (int kt = 0; kt < KT; ++kt) {
    const unsigned* p32 = (const unsigned*)(src + nl * strideH + kt * 32);
    union { unsigned u[8]; v16h h; } r;
#pragma unroll
    for (int p = 0; p < 8; ++p)
      r.u[p] = p32[(hf << 3) + p];
    a[kt] = r.h;
  }

  constexpr int STEP = (NT % 2 == 0) ? 2 : 1;
#pragma unroll
  for (int nt = 0; nt < NT; nt += STEP) {
    float bv0 = (nt * 16 + nl < nBound) ? bias[nt * 16 + nl] : 0.f;
    v8f acc0 = {bv0, bv0, bv0, bv0, bv0, bv0, bv0, bv0};
    v8f acc1;
    if constexpr (STEP == 2) {
      float bv1 = ((nt + 1) * 16 + nl < nBound) ? bias[(nt + 1) * 16 + nl] : 0.f;
      acc1 = (v8f){bv1, bv1, bv1, bv1, bv1, bv1, bv1, bv1};
    }
#pragma unroll
    for (int kt = 0; kt < KT; ++kt) {
      // issue both B-fragment loads first -> one clause, one wait, two WMMAs
      v16h b0 = ld_bfrag(fragW + ((size_t)(kt * NT + nt)) * 512 + lane * 16);
      v16h b1;
      if constexpr (STEP == 2)
        b1 = ld_bfrag(fragW + ((size_t)(kt * NT + nt + 1)) * 512 + lane * 16);
      if (kt + 1 < KT)
        __builtin_prefetch(fragW + ((size_t)((kt + 1) * NT + nt)) * 512 + lane * 16, 0, 3);
      acc0 = __builtin_amdgcn_wmma_f32_16x16x32_f16(
          false, a[kt], false, b0, (short)0, acc0, false, false);
      if constexpr (STEP == 2)
        acc1 = __builtin_amdgcn_wmma_f32_16x16x32_f16(
            false, a[kt], false, b1, (short)0, acc1, false, false);
    }
#pragma unroll
    for (int s = 0; s < STEP; ++s) {
      v8f acc = (s == 0) ? acc0 : acc1;
      if (doSilu) {
#pragma unroll
        for (int p = 0; p < 8; ++p) acc[p] = fast_silu(acc[p]);
      }
      if (dst) {
#pragma unroll
        for (int p = 0; p < 8; ++p)
          dst[(p + 8 * hf) * strideH + (nt + s) * 16 + nl] = (half_t)acc[p];
      } else {
        accOut[nt + s] = acc;
      }
    }
  }
}

// In-register LayerNorm over 128 features held as 8 C-fragments.
__device__ __forceinline__ void ln128(v8f accs[8], const float* g,
                                      const float* b, int lane) {
  const int nl = lane & 15;
  float mean[8], rstd[8];
#pragma unroll
  for (int p = 0; p < 8; ++p) {
    float s = 0.f, q = 0.f;
#pragma unroll
    for (int nt = 0; nt < 8; ++nt) { float t = accs[nt][p]; s += t; q += t * t; }
    for (int m = 1; m < 16; m <<= 1) {
      s += __shfl_xor(s, m, 32);
      q += __shfl_xor(q, m, 32);
    }
    float mu  = s * (1.f / 128.f);
    float var = q * (1.f / 128.f) - mu * mu;
    mean[p] = mu;
    rstd[p] = fast_rsqrt(var + 1e-5f);
  }
#pragma unroll
  for (int nt = 0; nt < 8; ++nt) {
    float gg = g[nt * 16 + nl], bb = b[nt * 16 + nl];
#pragma unroll
    for (int p = 0; p < 8; ++p)
      accs[nt][p] = (accs[nt][p] - mean[p]) * rstd[p] * gg + bb;
  }
}

// ---- utility kernels -------------------------------------------------------

__global__ void zero_kernel(float* aggT, float* aggE) {
  long t = (long)blockIdx.x * blockDim.x + threadIdx.x;
  if (t < (long)NN * 16)       aggT[t] = 0.f;
  else if (t < (long)NN * 144) aggE[t - (long)NN * 16] = 0.f;
}

// scatter edge_type to agg_t + produce f16 copy of edge_type
__global__ void scatter_et(const float* __restrict__ et,
                           const int* __restrict__ ej,
                           float* aggT, half_t* __restrict__ et16) {
  long t = (long)blockIdx.x * blockDim.x + threadIdx.x;
  if (t >= (long)EE * 16) return;
  long e = t >> 4; int d = (int)(t & 15);
  float v = et[t];
  et16[t] = (half_t)v;
  atomicAdd(aggT + (long)ej[e] * 16 + d, v);
}

// Repack f32 weight W[Kdim][Ndim] into f16 WMMA B fragments.
// B layout (wave32, 16-bit): lane = column N (nl), lane-half splits K by 16,
// k = kt*32 + hf*16 + p*2 + hh. Out-of-range -> 0 (K/N padding).
__global__ void prepack(const float* __restrict__ W, half_t* __restrict__ dst,
                        int Kdim, int Ndim, int KT, int NT) {
  int t = blockIdx.x * blockDim.x + threadIdx.x;
  if (t >= KT * NT * 32) return;
  int lane = t & 31, f = t >> 5;
  int kt = f / NT, nt = f % NT;
  int nl = lane & 15, hf = lane >> 4;
  int N = nt * 16 + nl;
  half_t* o = dst + (size_t)f * 512 + lane * 16;
#pragma unroll
  for (int p = 0; p < 8; ++p)
#pragma unroll
    for (int hh = 0; hh < 2; ++hh) {
      int k = kt * 32 + hf * 16 + p * 2 + hh;
      float v = (k < Kdim && N < Ndim) ? W[(size_t)k * Ndim + N] : 0.f;
      o[p * 2 + hh] = (half_t)v;
    }
}

// ---- node kernel: h = LN(MLP([node_h|agg_t])) + node_h ---------------------

struct NodeP {
  const float *node_h, *aggT;
  const half_t *fn0, *fn1, *fn2;
  const float *bn0, *bn1, *bn2, *gn, *bbn;
  float* out_h;
  half_t* h16;            // f16 copy of h for the edge kernel
};

#define NSTR 162           // 162 halves = 81 dwords (odd) -> conflict-free rows
#define NODE_WAVE_LDS 8192 // f32 staging tile 16x128 needs 8KB

__global__ __launch_bounds__(256) void node_kernel(NodeP p) {
  extern __shared__ char smem[];
  const int wave = threadIdx.x >> 5, lane = threadIdx.x & 31;
  const int nl = lane & 15, hf = lane >> 4;
  half_t* buf = (half_t*)(smem + wave * NODE_WAVE_LDS);
  const long base = (long)blockIdx.x * 128 + wave * 16;

  for (int m = 0; m < 16; ++m) {
    long n = base + m; if (n >= NN) n = NN - 1;
#pragma unroll
    for (int t = 0; t < 2; ++t) {
      int c2 = lane + t * 32;          // float-pair index, 64 pairs = 128 cols
      float2 v = ((const float2*)(p.node_h + n * 128))[c2];
      st_half2(buf + m * NSTR + 2 * c2, v.x, v.y);
    }
    buf[m * NSTR + 128 + lane] =
        (lane < 16) ? (half_t)p.aggT[n * 16 + lane] : (half_t)0.f;  // pad 144..159
  }

  v8f accs[8];
  mlp_layer<5, 8>(buf, NSTR, buf, p.fn0, p.bn0, 128, lane, true, nullptr);
  mlp_layer<4, 8>(buf, NSTR, buf, p.fn1, p.bn1, 128, lane, true, nullptr);
  mlp_layer<4, 8>(buf, NSTR, (half_t*)nullptr, p.fn2, p.bn2, 128, lane, false, accs);
  ln128(accs, p.gn, p.bbn, lane);

  float* fbuf = (float*)buf;   // A frags already consumed -> safe to alias
#pragma unroll
  for (int nt = 0; nt < 8; ++nt)
#pragma unroll
    for (int q = 0; q < 8; ++q) {
      long n = base + q + 8 * hf; if (n >= NN) n = NN - 1;
      fbuf[(q + 8 * hf) * 128 + nt * 16 + nl] =
          accs[nt][q] + p.node_h[n * 128 + nt * 16 + nl];   // residual
    }
  const float4* f4 = (const float4*)fbuf;
#pragma unroll
  for (int t = 0; t < 16; ++t) {
    int idx = t * 32 + lane, row = idx >> 5, c4 = idx & 31;
    long n = base + row;
    if (n < NN) ((float4*)p.out_h)[n * 32 + c4] = f4[idx];
  }
  // f16 copy of h (one conversion per node instead of one per edge endpoint)
#pragma unroll
  for (int t = 0; t < 32; ++t) {
    int idx = t * 32 + lane;           // 1024 half2 words
    int row = idx >> 6, cu = idx & 63;
    long n = base + row;
    if (n < NN) {
      float2 v = ((const float2*)fbuf)[idx];
      ((unsigned*)p.h16)[n * 64 + cu] = pack_half2(v.x, v.y);
    }
  }
}

// ---- edge mega-kernel ------------------------------------------------------

struct EdgeP {
  const half_t *h16, *et16;
  const float *x;
  const int *ei, *ej;
  const half_t *fe0, *fe1, *fe2, *fx0, *fx1, *fx2, *femb;
  const float *bh0, *bh1, *bh2, *bx0, *bx1, *bx2, *bemb;
  const float *gh, *bbh, *gx, *bbx;
  float *out_edge, *aggE;
};

#define ESTR 296            // 148 dwords: 16B-aligned rows, conflict-free A loads
#define EDGE_WAVE_LDS 9472  // 16*296*2 bytes (>= 8KB f32 staging tile)

__global__ __launch_bounds__(256) void edge_kernel(EdgeP p) {
  extern __shared__ char smem[];
  const int wave = threadIdx.x >> 5, lane = threadIdx.x & 31;
  const int nl = lane & 15, hf = lane >> 4;
  half_t* buf = (half_t*)(smem + wave * EDGE_WAVE_LDS);
  const long eBase = (long)blockIdx.x * 128 + wave * 16;

  // Stage [h_i | h_j | edge_type | 0pad] -> 16 x 288 f16, b128 copies only.
  // lanes 0..15 move h_i (16B chunk nl), lanes 16..31 move h_j.
  for (int m = 0; m < 16; ++m) {
    long e = eBase + m;
    const int vi = p.ei[e], vj = p.ej[e];
    const uint4* srcp = (const uint4*)(p.h16 + (long)(hf ? vj : vi) * 128);
    uint4 v = srcp[nl];
    *(uint4*)(buf + m * ESTR + hf * 128 + nl * 8) = v;
    if (lane < 4) {
      uint4 w{0, 0, 0, 0};
      if (lane < 2) w = ((const uint4*)(p.et16 + e * 16))[lane];
      *(uint4*)(buf + m * ESTR + 256 + lane * 8) = w;   // et + zero pad
    }
  }

  v8f accs[8];
  // MLP_h layer0: 272(+pad)->128, SiLU (A frags loaded before dst overwrite)
  mlp_layer<9, 8>(buf, ESTR, buf, p.fe0, p.bh0, 128, lane, true, nullptr);

  // Stage MLP_x input [disp(3)|dlen|0pad] at cols 128..159
  if (lane < 16) {
    long e = eBase + lane;
    int vi = p.ei[e], vj = p.ej[e];
    float d0 = p.x[(long)vj * 3 + 0] - p.x[(long)vi * 3 + 0];
    float d1 = p.x[(long)vj * 3 + 1] - p.x[(long)vi * 3 + 1];
    float d2 = p.x[(long)vj * 3 + 2] - p.x[(long)vi * 3 + 2];
    float dl = sqrtf(d0 * d0 + d1 * d1 + d2 * d2);
    half_t* r = buf + lane * ESTR + 128;
    st_half2(r + 0, d0, d1);
    st_half2(r + 2, d2, dl);
#pragma unroll
    for (int c = 4; c < 32; c += 2) st_half2(r + c, 0.f, 0.f);
  }

  mlp_layer<4, 8>(buf, ESTR, buf, p.fe1, p.bh1, 128, lane, true, nullptr);
  mlp_layer<4, 8>(buf, ESTR, (half_t*)nullptr, p.fe2, p.bh2, 128, lane, false, accs);
  ln128(accs, p.gh, p.bbh, lane);
#pragma unroll
  for (int nt = 0; nt < 8; ++nt)                  // out_h -> cols 0..127
#pragma unroll
    for (int q = 0; q < 8; ++q)
      buf[(q + 8 * hf) * ESTR + nt * 16 + nl] = (half_t)accs[nt][q];

  mlp_layer<1, 8>(buf + 128, ESTR, buf + 160, p.fx0, p.bx0, 128, lane, true, nullptr);
  mlp_layer<4, 8>(buf + 160, ESTR, buf + 160, p.fx1, p.bx1, 128, lane, true, nullptr);
  mlp_layer<4, 8>(buf + 160, ESTR, (half_t*)nullptr, p.fx2, p.bx2, 128, lane, false, accs);
  ln128(accs, p.gx, p.bbx, lane);
#pragma unroll
  for (int nt = 0; nt < 8; ++nt)                  // out_x -> cols 128..255
#pragma unroll
    for (int q = 0; q < 8; ++q)
      buf[(q + 8 * hf) * ESTR + 128 + nt * 16 + nl] = (half_t)accs[nt][q];

  // embed_out: [out_h|out_x](16x256) @ W(256x128) + b
  mlp_layer<8, 8>(buf, ESTR, (half_t*)nullptr, p.femb, p.bemb, 128, lane, false, accs);

  float* fbuf = (float*)buf;   // embed A frags already in registers
#pragma unroll
  for (int nt = 0; nt < 8; ++nt)
#pragma unroll
    for (int q = 0; q < 8; ++q)
      fbuf[(q + 8 * hf) * 128 + nt * 16 + nl] = accs[nt][q];

  // Coalesced edge_h writeout + fp32 scatter-add into agg_e (L2 atomics)
  const float4* f4 = (const float4*)fbuf;
#pragma unroll
  for (int t = 0; t < 16; ++t) {
    int idx = t * 32 + lane, row = idx >> 5, c4 = idx & 31;
    float4 v = f4[idx];
    long e = eBase + row;
    ((float4*)p.out_edge)[e * 32 + c4] = v;
    float* ag = p.aggE + (long)p.ej[e] * 128 + c4 * 4;
    atomicAdd(ag + 0, v.x); atomicAdd(ag + 1, v.y);
    atomicAdd(ag + 2, v.z); atomicAdd(ag + 3, v.w);
  }
}

// ---- pos kernel: x_out = LN3(MLP([x|agg_e])) + x ---------------------------

struct PosP {
  const float *x, *aggE;
  const half_t *fp0, *fp1, *fp2;
  const float *bp0, *bp1, *bp2, *gp, *bbp;
  float* out_x;
};

#define POS_WAVE_LDS 5184   // 16*162*2

__global__ __launch_bounds__(256) void pos_kernel(PosP p) {
  extern __shared__ char smem[];
  const int wave = threadIdx.x >> 5, lane = threadIdx.x & 31;
  const int nl = lane & 15, hf = lane >> 4;
  half_t* buf = (half_t*)(smem + wave * POS_WAVE_LDS);
  const long base = (long)blockIdx.x * 128 + wave * 16;

  for (int m = 0; m < 16; ++m) {
    long n = base + m; if (n >= NN) n = NN - 1;
#pragma unroll
    for (int t = 0; t < 5; ++t) {
      int c = t * 32 + lane;
      if (c < 160) {
        float v = 0.f;
        if (c < 3)        v = p.x[n * 3 + c];
        else if (c < 131) v = p.aggE[n * 128 + (c - 3)];
        buf[m * NSTR + c] = (half_t)v;
      }
    }
  }

  v8f accs[8];
  mlp_layer<5, 8>(buf, NSTR, buf, p.fp0, p.bp0, 128, lane, true, nullptr);
  mlp_layer<4, 8>(buf, NSTR, buf, p.fp1, p.bp1, 128, lane, true, nullptr);
  mlp_layer<4, 1>(buf, NSTR, (half_t*)nullptr, p.fp2, p.bp2, 3, lane, false, accs);

  float* fbuf = (float*)buf;
#pragma unroll
  for (int q = 0; q < 8; ++q) fbuf[(q + 8 * hf) * 16 + nl] = accs[0][q];

  if (lane < 16) {
    long n = base + lane;
    if (n < NN) {
      float y0 = fbuf[lane * 16 + 0], y1 = fbuf[lane * 16 + 1], y2 = fbuf[lane * 16 + 2];
      float mu = (y0 + y1 + y2) * (1.f / 3.f);
      float v0 = y0 - mu, v1 = y1 - mu, v2 = y2 - mu;
      float var = (v0 * v0 + v1 * v1 + v2 * v2) * (1.f / 3.f);
      float r = fast_rsqrt(var + 1e-5f);
      p.out_x[n * 3 + 0] = v0 * r * p.gp[0] + p.bbp[0] + p.x[n * 3 + 0];
      p.out_x[n * 3 + 1] = v1 * r * p.gp[1] + p.bbp[1] + p.x[n * 3 + 1];
      p.out_x[n * 3 + 2] = v2 * r * p.gp[2] + p.bbp[2] + p.x[n * 3 + 2];
    }
  }
}

// ---- host launch -----------------------------------------------------------
// d_in order (top-level dict insertion order, params pytree with sorted keys):
// 0 node_h, 1 x, 2 edge_index, 3 edge_type_h,
// 4/5 edge_ln_h b,g  6/7 edge_ln_x b,g
// 8..13 edge_mlp_h (W,b)x3   14..19 edge_mlp_x (W,b)x3
// 20/21 embed_out W,b   22/23 node_ln b,g   24..29 node_mlp (W,b)x3
// 30/31 pos_ln b,g   32..37 pos_mlp (W,b)x3

extern "C" void kernel_launch(void* const* d_in, const int* in_sizes, int n_in,
                              void* d_out, int out_size, void* d_ws, size_t ws_size,
                              hipStream_t stream) {
  (void)in_sizes; (void)n_in; (void)out_size;

  const float* node_h = (const float*)d_in[0];
  const float* x      = (const float*)d_in[1];
  const int*   ei     = (const int*)d_in[2];
  const int*   ej     = ei + EE;
  const float* et     = (const float*)d_in[3];
  const float *eln_h_b = (const float*)d_in[4],  *eln_h_g = (const float*)d_in[5];
  const float *eln_x_b = (const float*)d_in[6],  *eln_x_g = (const float*)d_in[7];
  const float *ehW0 = (const float*)d_in[8],  *eh_b0 = (const float*)d_in[9];
  const float *ehW1 = (const float*)d_in[10], *eh_b1 = (const float*)d_in[11];
  const float *ehW2 = (const float*)d_in[12], *eh_b2 = (const float*)d_in[13];
  const float *exW0 = (const float*)d_in[14], *ex_b0 = (const float*)d_in[15];
  const float *exW1 = (const float*)d_in[16], *ex_b1 = (const float*)d_in[17];
  const float *exW2 = (const float*)d_in[18], *ex_b2 = (const float*)d_in[19];
  const float *embW = (const float*)d_in[20], *emb_b = (const float*)d_in[21];
  const float *nln_b = (const float*)d_in[22], *nln_g = (const float*)d_in[23];
  const float *nW0 = (const float*)d_in[24], *n_b0 = (const float*)d_in[25];
  const float *nW1 = (const float*)d_in[26], *n_b1 = (const float*)d_in[27];
  const float *nW2 = (const float*)d_in[28], *n_b2 = (const float*)d_in[29];
  const float *pln_b = (const float*)d_in[30], *pln_g = (const float*)d_in[31];
  const float *pW0 = (const float*)d_in[32], *p_b0 = (const float*)d_in[33];
  const float *pW1 = (const float*)d_in[34], *p_b1 = (const float*)d_in[35];
  const float *pW2 = (const float*)d_in[36], *p_b2 = (const float*)d_in[37];

  float* out_edge = (float*)d_out;
  float* out_h    = out_edge + (size_t)EE * 128;
  float* out_x    = out_h + (size_t)NN * 128;

  // workspace layout (bytes):
  //   0        : aggT  N*16 f32        (1,280,000)
  //   1.28e6   : aggE  N*128 f32       (10,240,000)
  //   11.52e6  : h16   N*128 f16       (5,120,000)
  //   16.64e6  : et16  E*16  f16       (10,240,000)
  //   26.88e6  : weight fragments      (452 KiB)
  float*  aggT = (float*)d_ws;
  float*  aggE = aggT + (size_t)NN * 16;
  half_t* h16  = (half_t*)((char*)d_ws + 11520000);
  half_t* et16 = (half_t*)((char*)d_ws + 16640000);
  half_t* frag = (half_t*)((char*)d_ws + 26880000);
  half_t *fn0 = frag,            *fn1 = frag + 40 * 512,  *fn2 = frag + 72 * 512;
  half_t *fe0 = frag + 104*512,  *fe1 = frag + 176 * 512, *fe2 = frag + 208 * 512;
  half_t *fx0 = frag + 240*512,  *fx1 = frag + 248 * 512, *fx2 = frag + 280 * 512;
  half_t *femb = frag + 312*512;
  half_t *fp0 = frag + 376*512,  *fp1 = frag + 416 * 512, *fp2 = frag + 448 * 512;
  if (ws_size < 26880000 + 452 * 1024) return;   // ~27.4MB scratch needed

  zero_kernel<<<((long)NN * 144 + 255) / 256, 256, 0, stream>>>(aggT, aggE);

  auto pk = [&](const float* W, half_t* dst, int K, int Nd, int KT, int NT) {
    int th = KT * NT * 32;
    prepack<<<(th + 255) / 256, 256, 0, stream>>>(W, dst, K, Nd, KT, NT);
  };
  pk(nW0, fn0, 144, 128, 5, 8); pk(nW1, fn1, 128, 128, 4, 8); pk(nW2, fn2, 128, 128, 4, 8);
  pk(ehW0, fe0, 272, 128, 9, 8); pk(ehW1, fe1, 128, 128, 4, 8); pk(ehW2, fe2, 128, 128, 4, 8);
  pk(exW0, fx0, 4, 128, 1, 8);   pk(exW1, fx1, 128, 128, 4, 8); pk(exW2, fx2, 128, 128, 4, 8);
  pk(embW, femb, 256, 128, 8, 8);
  pk(pW0, fp0, 131, 128, 5, 8);  pk(pW1, fp1, 128, 128, 4, 8);  pk(pW2, fp2, 128, 3, 4, 1);

  scatter_et<<<((long)EE * 16 + 255) / 256, 256, 0, stream>>>(et, ej, aggT, et16);

  NodeP np{node_h, aggT, fn0, fn1, fn2, n_b0, n_b1, n_b2, nln_g, nln_b, out_h, h16};
  (void)hipFuncSetAttribute((const void*)node_kernel,
                            hipFuncAttributeMaxDynamicSharedMemorySize,
                            8 * NODE_WAVE_LDS);
  node_kernel<<<(NN + 127) / 128, 256, 8 * NODE_WAVE_LDS, stream>>>(np);

  EdgeP ep{h16, et16, x, ei, ej, fe0, fe1, fe2, fx0, fx1, fx2, femb,
           eh_b0, eh_b1, eh_b2, ex_b0, ex_b1, ex_b2, emb_b,
           eln_h_g, eln_h_b, eln_x_g, eln_x_b, out_edge, aggE};
  (void)hipFuncSetAttribute((const void*)edge_kernel,
                            hipFuncAttributeMaxDynamicSharedMemorySize,
                            8 * EDGE_WAVE_LDS);
  edge_kernel<<<EE / 128, 256, 8 * EDGE_WAVE_LDS, stream>>>(ep);

  PosP pp{x, aggE, fp0, fp1, fp2, p_b0, p_b1, p_b2, pln_g, pln_b, out_x};
  (void)hipFuncSetAttribute((const void*)pos_kernel,
                            hipFuncAttributeMaxDynamicSharedMemorySize,
                            8 * POS_WAVE_LDS);
  pos_kernel<<<(NN + 127) / 128, 256, 8 * POS_WAVE_LDS, stream>>>(pp);
}
